// ExtractorHead_50397146251649
// MI455X (gfx1250) — matrix operand order repaired and checked by reading, not compiled, not run on hardware
//
#include <hip/hip_runtime.h>

#define HIDDEN 128
#define NNODES 16384
#define EDGEF  20
#define CUTOFF 6.0f
#define PI_F   3.14159265358979323846f

typedef __attribute__((ext_vector_type(16))) _Float16 v16h;
typedef __attribute__((ext_vector_type(8)))  float    v8f;

__device__ __forceinline__ float fast_silu(float x) {
  // x * sigmoid(x) with v_rcp_f32 instead of IEEE division
  return x * __builtin_amdgcn_rcpf(1.0f + __expf(-x));
}

// ---------------------------------------------------------------------------
// Weight pre-swizzle: W[K,Nc] f32 (row-major) -> Wf f16 in fragment-linear
// order: Wf[((k0*Nc + n)*32 + e)] = W[k0*32 + e][n],  k0 = k-chunk index.
// A lane's 16-half B fragment (col n, K = kb2..kb2+15) is then contiguous
// and 32B-aligned -> two global_load_b128 per fragment.
// ---------------------------------------------------------------------------
__global__ void convert_w_kernel(const float* __restrict__ W,
                                 _Float16* __restrict__ Wf, int K, int Nc)
{
  int i = blockIdx.x * blockDim.x + threadIdx.x;
  if (i >= K * Nc) return;
  int e  = i & 31;
  int r  = i >> 5;
  int n  = r % Nc;
  int k0 = r / Nc;
  Wf[i] = (_Float16)W[(size_t)(k0 * 32 + e) * Nc + n];
}

// ---------------------------------------------------------------------------
// Generic WMMA GEMM: C[M,Nc] = act(A[M,K] @ W[K,Nc] + bias[Nc])
// One wave per 16x64 output strip; A fragment loaded once per k-chunk
// (4x b128 + cvt_pk) and reused by 4 v_wmma_f32_16x16x32_f16; B fragments
// are contiguous b128 loads from the pre-swizzled f16 weights.
// A/B fragments use contiguous k order: k = 16*half + s.
// ---------------------------------------------------------------------------
template <int ACT>
__global__ void __launch_bounds__(32) wmma_gemm_kernel(
    const float* __restrict__ A, const _Float16* __restrict__ Wf,
    const float* __restrict__ bias, float* __restrict__ C,
    int M, int K, int Nc)
{
  const int lane = threadIdx.x;
  const int half = lane >> 4;
  const int l15  = lane & 15;
  const int tm = blockIdx.x;           // 16-row strip
  const int nbase = blockIdx.y * 64;   // 64-col strip

  v8f acc0 = {0.f,0.f,0.f,0.f,0.f,0.f,0.f,0.f};
  v8f acc1 = acc0, acc2 = acc0, acc3 = acc0;

  const float* Arow = A + (size_t)(tm * 16 + l15) * K + half * 16;
  // lane's base into swizzled weights: chunk k0, col nbase+l15, e = half*16
  const _Float16* Wl = Wf + ((size_t)nbase + l15) * 32 + half * 16;

  for (int k0 = 0; k0 < K; k0 += 32) {
    // A fragment: 4x b128 loads, convert to f16 once, reuse across 4 WMMAs
    const float4 p0 = *(const float4*)(Arow + k0);
    const float4 p1 = *(const float4*)(Arow + k0 + 4);
    const float4 p2 = *(const float4*)(Arow + k0 + 8);
    const float4 p3 = *(const float4*)(Arow + k0 + 12);
    v16h a;
    a[0]=(_Float16)p0.x; a[1]=(_Float16)p0.y; a[2]=(_Float16)p0.z; a[3]=(_Float16)p0.w;
    a[4]=(_Float16)p1.x; a[5]=(_Float16)p1.y; a[6]=(_Float16)p1.z; a[7]=(_Float16)p1.w;
    a[8]=(_Float16)p2.x; a[9]=(_Float16)p2.y; a[10]=(_Float16)p2.z; a[11]=(_Float16)p2.w;
    a[12]=(_Float16)p3.x; a[13]=(_Float16)p3.y; a[14]=(_Float16)p3.z; a[15]=(_Float16)p3.w;

    // B fragments: contiguous 32B (2x b128) per 16-col subtile
    const _Float16* wb = Wl + ((size_t)(k0 >> 5) * Nc) * 32;
    v16h b0 = *(const v16h*)(wb);
    v16h b1 = *(const v16h*)(wb + 16 * 32);
    v16h b2 = *(const v16h*)(wb + 32 * 32);
    v16h b3 = *(const v16h*)(wb + 48 * 32);

    acc0 = __builtin_amdgcn_wmma_f32_16x16x32_f16(false, a, false, b0, (short)0, acc0, false, false);
    acc1 = __builtin_amdgcn_wmma_f32_16x16x32_f16(false, a, false, b1, (short)0, acc1, false, false);
    acc2 = __builtin_amdgcn_wmma_f32_16x16x32_f16(false, a, false, b2, (short)0, acc2, false, false);
    acc3 = __builtin_amdgcn_wmma_f32_16x16x32_f16(false, a, false, b3, (short)0, acc3, false, false);
  }

  // Epilogue: hoisted base pointer, stride Nc per row; no per-element branch.
  float* crow = C + (size_t)(tm * 16 + half * 8) * Nc + nbase + l15;
  const float* bb = bias + nbase + l15;
#pragma unroll
  for (int t = 0; t < 4; ++t) {
    const v8f acc = (t == 0) ? acc0 : (t == 1) ? acc1 : (t == 2) ? acc2 : acc3;
    const float bn = bb[t * 16];
    float* cp = crow + t * 16;
#pragma unroll
    for (int r = 0; r < 8; ++r) {
      float x = acc[r] + bn;
      if (ACT) x = fast_silu(x);
      cp[(size_t)r * Nc] = x;
    }
  }
}

// ---------------------------------------------------------------------------
// s[i] = embed[z[node]][h]; v = 0
// ---------------------------------------------------------------------------
__global__ void init_kernel(const int* __restrict__ z,
                            const float* __restrict__ embed,
                            float* __restrict__ s, float* __restrict__ v, int n)
{
  int i = blockIdx.x * blockDim.x + threadIdx.x;
  if (i >= n) return;
  v[i] = 0.f;
  if (i < NNODES * HIDDEN) {
    int node = i >> 7, h = i & 127;
    s[i] = embed[(size_t)z[node] * HIDDEN + h];
  }
}

__global__ void edge_dist_kernel(const float* __restrict__ pos,
                                 const int* __restrict__ ei, int E,
                                 float* __restrict__ dist_out)
{
  int e = blockIdx.x * blockDim.x + threadIdx.x;
  if (e >= E) return;
  int r = ei[e], c = ei[E + e];
  float dx = pos[3 * r] - pos[3 * c];
  float dy = pos[3 * r + 1] - pos[3 * c + 1];
  float dz = pos[3 * r + 2] - pos[3 * c + 2];
  dist_out[e] = sqrtf(dx * dx + dy * dy + dz * dz);
}

// ---------------------------------------------------------------------------
// Edge message pass: one wave per edge. Reads vOld[col]/so[col], atomically
// accumulates msg_s into s[row] and msg_v into vNew[row]. f_W cached in LDS.
// ---------------------------------------------------------------------------
__global__ void __launch_bounds__(128) edge_message_kernel(
    const float* __restrict__ pos, const int* __restrict__ ei, int E,
    const float* __restrict__ fW, const float* __restrict__ fb,
    const float* __restrict__ so,   // [N,384] scalar_out
    const float* __restrict__ vOld, // [N,3,128]
    float* __restrict__ sAcc,       // [N,128]
    float* __restrict__ vNew)       // [N,3,128]
{
  __shared__ float lfW[EDGEF * 3 * HIDDEN]; // 20*384 = 30720 B
  __shared__ float lfb[3 * HIDDEN];
  for (int i = threadIdx.x; i < EDGEF * 3 * HIDDEN; i += blockDim.x) lfW[i] = fW[i];
  for (int i = threadIdx.x; i < 3 * HIDDEN; i += blockDim.x) lfb[i] = fb[i];
  __syncthreads();

  const int lane   = threadIdx.x & 31;
  const int wave   = blockIdx.x * (blockDim.x >> 5) + (threadIdx.x >> 5);
  const int nwaves = gridDim.x * (blockDim.x >> 5);

  for (int e = wave; e < E; e += nwaves) {
    int r = ei[e], c = ei[E + e];
    float dx = pos[3 * r] - pos[3 * c];
    float dy = pos[3 * r + 1] - pos[3 * c + 1];
    float dz = pos[3 * r + 2] - pos[3 * c + 2];
    float d   = sqrtf(dx * dx + dy * dy + dz * dz);
    float inv = __builtin_amdgcn_rcpf(d);
    float ux = dx * inv, uy = dy * inv, uz = dz * inv;
    float cut = (d < CUTOFF) ? 0.5f * (__cosf(PI_F * d * (1.0f / CUTOFF)) + 1.0f) : 0.f;

    float sinc[EDGEF];
#pragma unroll
    for (int k = 0; k < EDGEF; ++k)
      sinc[k] = __sinf((float)(k + 1) * (PI_F / CUTOFF) * d) * inv;

    const float* soc = so   + (size_t)c * 384;
    const float* vc  = vOld + (size_t)c * 384;

#pragma unroll
    for (int hh = 0; hh < 4; ++hh) {
      int h = hh * 32 + lane;
      float f0 = lfb[h], f1 = lfb[h + 128], f2 = lfb[h + 256];
#pragma unroll
      for (int k = 0; k < EDGEF; ++k) {
        float sk = sinc[k];
        f0 += sk * lfW[k * 384 + h];
        f1 += sk * lfW[k * 384 + h + 128];
        f2 += sk * lfW[k * 384 + h + 256];
      }
      float gate_sv = f0 * cut * soc[h];
      float gate_ev = f1 * cut * soc[h + 128];
      float msg_s   = f2 * cut * soc[h + 256];

      atomicAdd(&sAcc[(size_t)r * 128 + h], msg_s);
      float v0 = vc[h], v1 = vc[128 + h], v2 = vc[256 + h];
      atomicAdd(&vNew[(size_t)r * 384 + h],       v0 * gate_sv + gate_ev * ux);
      atomicAdd(&vNew[(size_t)r * 384 + 128 + h], v1 * gate_sv + gate_ev * uy);
      atomicAdd(&vNew[(size_t)r * 384 + 256 + h], v2 * gate_sv + gate_ev * uz);
    }
  }
}

// ---------------------------------------------------------------------------
// mlp_in = concat(||Vv||_comp, s)
// ---------------------------------------------------------------------------
__global__ void vvnorm_kernel(const float* __restrict__ Vv,
                              const float* __restrict__ s,
                              float* __restrict__ mlp_in, int n)
{
  int i = blockIdx.x * blockDim.x + threadIdx.x;
  if (i >= n) return;
  int node = i >> 7, h = i & 127;
  const float* p = Vv + (size_t)node * 384;
  float a = p[h], b = p[128 + h], c = p[256 + h];
  mlp_in[(size_t)node * 256 + h]       = sqrtf(a * a + b * b + c * c);
  mlp_in[(size_t)node * 256 + 128 + h] = s[i];
}

// ---------------------------------------------------------------------------
// s += a_sv * <Uv,Vv> + a_ss;  v += a_vv * Uv
// ---------------------------------------------------------------------------
__global__ void update_kernel(const float* __restrict__ mo,
                              const float* __restrict__ Uv,
                              const float* __restrict__ Vv,
                              float* __restrict__ s, float* __restrict__ v, int n)
{
  int i = blockIdx.x * blockDim.x + threadIdx.x;
  if (i >= n) return;
  int node = i >> 7, h = i & 127;
  size_t b3 = (size_t)node * 384;
  float avv = mo[b3 + h], asv = mo[b3 + 128 + h], ass = mo[b3 + 256 + h];
  float u0 = Uv[b3 + h], u1 = Uv[b3 + 128 + h], u2 = Uv[b3 + 256 + h];
  float w0 = Vv[b3 + h], w1 = Vv[b3 + 128 + h], w2 = Vv[b3 + 256 + h];
  float inner = u0 * w0 + u1 * w1 + u2 * w2;
  s[i] += asv * inner + ass;
  v[b3 + h]       += avv * u0;
  v[b3 + 128 + h] += avv * u1;
  v[b3 + 256 + h] += avv * u2;
}

// ---------------------------------------------------------------------------
extern "C" void kernel_launch(void* const* d_in, const int* in_sizes, int n_in,
                              void* d_out, int out_size, void* d_ws, size_t ws_size,
                              hipStream_t stream)
{
  (void)out_size; (void)ws_size;
  const int*   z   = (const int*)d_in[0];
  const float* pos = (const float*)d_in[1];
  const int*   ei  = (const int*)d_in[2];
  const int    E   = in_sizes[2] / 2;

  // Parameters: either one concatenated buffer (jax tree-flatten, sorted keys)
  // or individual leaves in the same order.
  size_t off = 0; int idx = 3;
  const float* pbase = (n_in == 4) ? (const float*)d_in[3] : nullptr;
  auto nxt = [&](size_t cnt) -> const float* {
    if (pbase) { const float* p = pbase + off; off += cnt; return p; }
    return (const float*)d_in[idx++];
  };
  // order: embed | head{W1,W2,b1,b2} | msg[0..1]{f_W,f_b,m1_W,m1_b,m2_W,m2_b}
  //        | upd[0..1]{U_W,U_b,V_W,V_b,u1_W,u1_b,u2_W,u2_b}
  const float* embed = nxt(119 * 128);
  const float* hW1 = nxt(128 * 128);
  const float* hW2 = nxt(128 * 128);
  const float* hb1 = nxt(128);
  const float* hb2 = nxt(128);
  struct Msg { const float *fW, *fb, *m1W, *m1b, *m2W, *m2b; } msg[2];
  for (int l = 0; l < 2; ++l) {
    msg[l].fW  = nxt(20 * 384); msg[l].fb  = nxt(384);
    msg[l].m1W = nxt(128 * 128); msg[l].m1b = nxt(128);
    msg[l].m2W = nxt(128 * 384); msg[l].m2b = nxt(384);
  }
  struct Upd { const float *UW, *Ub, *VW, *Vb, *u1W, *u1b, *u2W, *u2b; } upd[2];
  for (int l = 0; l < 2; ++l) {
    upd[l].UW  = nxt(128 * 128); upd[l].Ub  = nxt(128);
    upd[l].VW  = nxt(128 * 128); upd[l].Vb  = nxt(128);
    upd[l].u1W = nxt(256 * 128); upd[l].u1b = nxt(128);
    upd[l].u2W = nxt(128 * 384); upd[l].u2b = nxt(384);
  }

  // Workspace layout (floats, then f16 swizzled weights)
  float* ws = (float*)d_ws;
  float* s   = ws; ws += (size_t)NNODES * 128;
  float* vA  = ws; ws += (size_t)NNODES * 384;
  float* vB  = ws; ws += (size_t)NNODES * 384;
  float* t1  = ws; ws += (size_t)NNODES * 128;  // MLP hidden scratch
  float* so  = ws; ws += (size_t)NNODES * 384;  // scalar_out / mlp_out
  float* Uv  = ws; ws += (size_t)NNODES * 384;
  float* Vv  = ws; ws += (size_t)NNODES * 384;
  float* mi  = ws; ws += (size_t)NNODES * 256;  // mlp_in

  _Float16* wf = (_Float16*)ws;
  auto walloc = [&](size_t cnt) { _Float16* p = wf; wf += cnt; return p; };
  _Float16* c_m1W[2], *c_m2W[2], *c_UW[2], *c_VW[2], *c_u1W[2], *c_u2W[2];
  for (int l = 0; l < 2; ++l) {
    c_m1W[l] = walloc(128 * 128);
    c_m2W[l] = walloc(128 * 384);
    c_UW[l]  = walloc(128 * 128);
    c_VW[l]  = walloc(128 * 128);
    c_u1W[l] = walloc(256 * 128);
    c_u2W[l] = walloc(128 * 384);
  }
  _Float16* c_hW1 = walloc(128 * 128);
  _Float16* c_hW2 = walloc(128 * 128);

  // Pre-swizzle all weights to f16 fragment-linear layout (~0.79 MB total)
  auto conv = [&](const float* W, _Float16* Wc, int K, int Nc) {
    convert_w_kernel<<<(K * Nc + 255) / 256, 256, 0, stream>>>(W, Wc, K, Nc);
  };
  for (int l = 0; l < 2; ++l) {
    conv(msg[l].m1W, c_m1W[l], 128, 128);
    conv(msg[l].m2W, c_m2W[l], 128, 384);
    conv(upd[l].UW,  c_UW[l],  128, 128);
    conv(upd[l].VW,  c_VW[l],  128, 128);
    conv(upd[l].u1W, c_u1W[l], 256, 128);
    conv(upd[l].u2W, c_u2W[l], 128, 384);
  }
  conv(hW1, c_hW1, 128, 128);
  conv(hW2, c_hW2, 128, 128);

  // Output layout: out [N,128] | pos [N,3] | edge_index [2,E] (bit-copied) | dist [E]
  float* out      = (float*)d_out;
  float* out_pos  = out + (size_t)NNODES * 128;
  float* out_ei   = out_pos + (size_t)NNODES * 3;
  float* out_dist = out_ei + (size_t)2 * E;

  hipMemcpyAsync(out_pos, pos, (size_t)NNODES * 3 * sizeof(float),
                 hipMemcpyDeviceToDevice, stream);
  hipMemcpyAsync(out_ei, ei, (size_t)2 * E * sizeof(int),
                 hipMemcpyDeviceToDevice, stream);
  edge_dist_kernel<<<(E + 255) / 256, 256, 0, stream>>>(pos, ei, E, out_dist);

  init_kernel<<<(NNODES * 384 + 255) / 256, 256, 0, stream>>>(z, embed, s, vA,
                                                              NNODES * 384);

  float* vcur = vA;
  float* vnext = vB;
  for (int l = 0; l < 2; ++l) {
    // scalar message MLP: so = silu(s@m1)@m2
    wmma_gemm_kernel<1><<<dim3(NNODES / 16, 2), 32, 0, stream>>>(
        s, c_m1W[l], msg[l].m1b, t1, NNODES, 128, 128);
    wmma_gemm_kernel<0><<<dim3(NNODES / 16, 6), 32, 0, stream>>>(
        t1, c_m2W[l], msg[l].m2b, so, NNODES, 128, 384);
    // double-buffer v for the scatter (read old, accumulate into copy)
    hipMemcpyAsync(vnext, vcur, (size_t)NNODES * 384 * sizeof(float),
                   hipMemcpyDeviceToDevice, stream);
    edge_message_kernel<<<1024, 128, 0, stream>>>(pos, ei, E, msg[l].fW,
                                                  msg[l].fb, so, vcur, s, vnext);
    { float* t = vcur; vcur = vnext; vnext = t; }
    // update block
    wmma_gemm_kernel<0><<<dim3(3 * NNODES / 16, 2), 32, 0, stream>>>(
        vcur, c_UW[l], upd[l].Ub, Uv, 3 * NNODES, 128, 128);
    wmma_gemm_kernel<0><<<dim3(3 * NNODES / 16, 2), 32, 0, stream>>>(
        vcur, c_VW[l], upd[l].Vb, Vv, 3 * NNODES, 128, 128);
    vvnorm_kernel<<<(NNODES * 128 + 255) / 256, 256, 0, stream>>>(
        Vv, s, mi, NNODES * 128);
    wmma_gemm_kernel<1><<<dim3(NNODES / 16, 2), 32, 0, stream>>>(
        mi, c_u1W[l], upd[l].u1b, t1, NNODES, 256, 128);
    wmma_gemm_kernel<0><<<dim3(NNODES / 16, 6), 32, 0, stream>>>(
        t1, c_u2W[l], upd[l].u2b, so, NNODES, 128, 384);
    update_kernel<<<(NNODES * 128 + 255) / 256, 256, 0, stream>>>(
        so, Uv, Vv, s, vcur, NNODES * 128);
  }

  // head: out = silu(s@W1+b1)@W2+b2
  wmma_gemm_kernel<1><<<dim3(NNODES / 16, 2), 32, 0, stream>>>(
      s, c_hW1, hb1, t1, NNODES, 128, 128);
  wmma_gemm_kernel<0><<<dim3(NNODES / 16, 2), 32, 0, stream>>>(
      t1, c_hW2, hb2, out, NNODES, 128, 128);
}
